// MalwareGNN_52158082842830
// MI455X (gfx1250) — compile-verified, matrix operand order checked by reading, not run on hardware
//
#include <hip/hip_runtime.h>
#include <hip/hip_fp16.h>
#include <math.h>
#include <stdint.h>

typedef __attribute__((ext_vector_type(16))) _Float16 v16h;
typedef __attribute__((ext_vector_type(8)))  float    v8f;

#define NN 100000      // nodes
#define NE 400000      // edges (without self loops)
#define EP 500000      // edges + self loops
#define NG 128         // graphs
#define NEG_INF_ORD 0x007FFFFFu   // f2o(-inf)

__device__ __forceinline__ unsigned f2o(float f) {
  unsigned u = __float_as_uint(f);
  return (u & 0x80000000u) ? ~u : (u | 0x80000000u);
}
__device__ __forceinline__ float o2f(unsigned u) {
  return __uint_as_float((u & 0x80000000u) ? (u & 0x7fffffffu) : ~u);
}

// gfx1250 async global->LDS copy (ASYNCcnt-tracked), per-lane 16B
__device__ __forceinline__ void async_copy_b128(unsigned lds_off, const void* gaddr) {
  asm volatile("global_load_async_to_lds_b128 %0, %1, off"
               :: "v"(lds_off), "v"((unsigned long long)(uintptr_t)gaddr)
               : "memory");
}
__device__ __forceinline__ void wait_async0() {
  asm volatile("s_wait_asynccnt 0x0" ::: "memory");
}

// ---------------------------------------------------------------------------
// WMMA GEMM:  C[M x Nout] = A[M x K](f16) @ W[K x Nout](f16) + bias, opt tanh
// 256 threads = 8 wave32; each wave owns a 16x16 C tile; block tile 128x16.
// K stepped by 64: A tile staged via GLOBAL_LOAD_ASYNC_TO_LDS_B128;
// two v_wmma_f32_16x16x32_f16 per barrier round.
// ---------------------------------------------------------------------------
__global__ void __launch_bounds__(256)
gemm_wmma_k(const _Float16* __restrict__ A, const _Float16* __restrict__ W,
            const float* __restrict__ bias, float* __restrict__ C,
            int M, int K, int Nout, int act) {
  __shared__ _Float16 ldsA[128 * 64];   // [m][k]   16 KB
  __shared__ _Float16 ldsB[16 * 64];    // [n][k]    2 KB (transposed)
  const int tid   = threadIdx.x;
  const int tileM = blockIdx.x * 128;
  const int n0    = blockIdx.y * 16;
  const int wave  = tid >> 5;
  const int lane  = tid & 31;
  const int g     = lane >> 4;       // lane group 0/1
  const int ln    = lane & 15;       // 0..15
  const int mrow  = wave * 16 + ln;  // A-fragment row within 128-tile

  // per-thread fixed staging assignments (straight-line, no divergent loops)
  // A: 1024 x 16B chunks (128 rows x 8 chunks); this thread owns 4 of them.
  const int ac0 = tid, ac1 = tid + 256, ac2 = tid + 512, ac3 = tid + 768;
  int gr0 = tileM + (ac0 >> 3); if (gr0 >= M) gr0 = M - 1;   // clamp: garbage rows
  int gr1 = tileM + (ac1 >> 3); if (gr1 >= M) gr1 = M - 1;   // feed only C rows
  int gr2 = tileM + (ac2 >> 3); if (gr2 >= M) gr2 = M - 1;   // that are never
  int gr3 = tileM + (ac3 >> 3); if (gr3 >= M) gr3 = M - 1;   // stored
  const int as0 = ac0 & 7, as1 = ac1 & 7, as2 = ac2 & 7, as3 = ac3 & 7;
  const unsigned la0 = (unsigned)(uintptr_t)&ldsA[ac0 * 8];
  const unsigned la1 = (unsigned)(uintptr_t)&ldsA[ac1 * 8];
  const unsigned la2 = (unsigned)(uintptr_t)&ldsA[ac2 * 8];
  const unsigned la3 = (unsigned)(uintptr_t)&ldsA[ac3 * 8];
  // B: 512 n-pair loads (64 k rows x 8 pairs); this thread owns 2.
  const int bp0 = tid, bp1 = tid + 256;
  const int bk0 = bp0 >> 3, bn0 = (bp0 & 7) * 2;
  const int bk1 = bp1 >> 3, bn1 = (bp1 & 7) * 2;

  v8f acc = {};

  for (int k0 = 0; k0 < K; k0 += 64) {
    __syncthreads();
    // A tile: async global -> LDS, no VGPR round trip
    async_copy_b128(la0, A + (size_t)gr0 * K + k0 + as0 * 8);
    async_copy_b128(la1, A + (size_t)gr1 * K + k0 + as1 * 8);
    async_copy_b128(la2, A + (size_t)gr2 * K + k0 + as2 * 8);
    async_copy_b128(la3, A + (size_t)gr3 * K + k0 + as3 * 8);
    // B tile: load two halves (consecutive n) per uint, scatter transposed
    {
      unsigned w0 = *(const unsigned*)(W + (size_t)(k0 + bk0) * Nout + n0 + bn0);
      unsigned w1 = *(const unsigned*)(W + (size_t)(k0 + bk1) * Nout + n0 + bn1);
      ldsB[(bn0    ) * 64 + bk0] = (_Float16)__ushort_as_half((unsigned short)(w0 & 0xffffu));
      ldsB[(bn0 + 1) * 64 + bk0] = (_Float16)__ushort_as_half((unsigned short)(w0 >> 16));
      ldsB[(bn1    ) * 64 + bk1] = (_Float16)__ushort_as_half((unsigned short)(w1 & 0xffffu));
      ldsB[(bn1 + 1) * 64 + bk1] = (_Float16)__ushort_as_half((unsigned short)(w1 >> 16));
    }
    wait_async0();          // this wave's async copies landed in LDS
    __syncthreads();        // all waves' tiles visible

    union { uint4 q[2]; v16h v; } a0, a1, b0, b1;
    const uint4* A4 = (const uint4*)ldsA;   // row stride 8 (64 halves)
    const uint4* B4 = (const uint4*)ldsB;
    a0.q[0] = A4[mrow * 8 + g];      a0.q[1] = A4[mrow * 8 + 2 + g];
    a1.q[0] = A4[mrow * 8 + 4 + g];  a1.q[1] = A4[mrow * 8 + 6 + g];
    b0.q[0] = B4[ln * 8 + g];        b0.q[1] = B4[ln * 8 + 2 + g];
    b1.q[0] = B4[ln * 8 + 4 + g];    b1.q[1] = B4[ln * 8 + 6 + g];

    acc = __builtin_amdgcn_wmma_f32_16x16x32_f16(false, a0.v, false, b0.v,
                                                 (short)0, acc, false, false);
    acc = __builtin_amdgcn_wmma_f32_16x16x32_f16(false, a1.v, false, b1.v,
                                                 (short)0, acc, false, false);
  }

  // C layout: VGPR r -> row r (lanes 0-15) / row 8+r (lanes 16-31), col = ln
  const int col = n0 + ln;
  const float bv = bias ? bias[col] : 0.0f;
  for (int r = 0; r < 8; ++r) {
    int row = tileM + wave * 16 + (g ? (r + 8) : r);
    if (row < M) {
      float v = acc[r] + bv;
      if (act == 1) v = tanhf(v);
      C[(size_t)row * Nout + col] = v;
    }
  }
}

// ---------------------------------------------------------------------------
// small utility kernels
// ---------------------------------------------------------------------------
__global__ void cvt_f16_k(const float* __restrict__ in, _Float16* __restrict__ out, int n) {
  int i = blockIdx.x * blockDim.x + threadIdx.x;
  if (i < n) out[i] = (_Float16)in[i];
}

__global__ void deg_init_k(float* deg) {
  int i = blockIdx.x * blockDim.x + threadIdx.x;
  if (i < NN) deg[i] = 1.0f;   // self loop
}
__global__ void deg_acc_k(const int* __restrict__ ei, float* deg) {
  int e = blockIdx.x * blockDim.x + threadIdx.x;
  if (e < NE) atomicAdd(&deg[ei[NE + e]], 1.0f);
}
__global__ void dis_k(const float* __restrict__ deg, float* dis) {
  int i = blockIdx.x * blockDim.x + threadIdx.x;
  if (i < NN) dis[i] = rsqrtf(fmaxf(deg[i], 1.0f));
}

__global__ void seg_init_k(unsigned* segmax, float* segsum) {
  int i = blockIdx.x * blockDim.x + threadIdx.x;
  if (i < NN * 4) { segmax[i] = NEG_INF_ORD; segsum[i] = 0.0f; }
}

// pass 1: e_h = dot(leaky(xl[src]+xr[dst]), att_h); segment max over dst
__global__ void gat_e1_k(const int* __restrict__ ei, const float* __restrict__ xl,
                         const float* __restrict__ xr, const float* __restrict__ att,
                         float* __restrict__ ebuf, unsigned* segmax, int dout) {
  int idx = blockIdx.x * blockDim.x + threadIdx.x;
  if (idx >= EP * 4) return;
  int e = idx >> 2, h = idx & 3;
  int s, d;
  if (e < NE) { s = ei[e]; d = ei[NE + e]; } else { s = d = e - NE; }
  int dph = dout >> 2;
  const float* pa = att + h * dph;
  const float* pl = xl + (size_t)s * dout + h * dph;
  const float* pr = xr + (size_t)d * dout + h * dph;
  float acc = 0.0f;
  for (int j = 0; j < dph; ++j) {
    float m = pl[j] + pr[j];
    m = m > 0.0f ? m : 0.2f * m;          // leaky_relu 0.2
    acc += m * pa[j];
  }
  ebuf[idx] = acc;
  atomicMax(&segmax[d * 4 + h], f2o(acc));
}

// pass 2: ez = exp(e - max[dst]); segment sum
__global__ void gat_e2_k(const int* __restrict__ ei, float* __restrict__ ebuf,
                         const unsigned* __restrict__ segmax, float* segsum) {
  int idx = blockIdx.x * blockDim.x + threadIdx.x;
  if (idx >= EP * 4) return;
  int e = idx >> 2, h = idx & 3;
  int d = (e < NE) ? ei[NE + e] : (e - NE);
  float ez = expf(ebuf[idx] - o2f(segmax[d * 4 + h]));
  ebuf[idx] = ez;
  atomicAdd(&segsum[d * 4 + h], ez);
}

// pass 3: acc[dst,f] += alpha*xl[src,f] + norm*hp[src,f]
__global__ void gat_e3_k(const int* __restrict__ ei, const float* __restrict__ ebuf,
                         const float* __restrict__ segsum, const float* __restrict__ xl,
                         const float* __restrict__ hp, const float* __restrict__ dis,
                         float* accb, int dout, int dshift) {
  int idx = blockIdx.x * blockDim.x + threadIdx.x;
  if (idx >= (EP << dshift)) return;
  int e = idx >> dshift, f = idx & (dout - 1);
  int s, d;
  if (e < NE) { s = ei[e]; d = ei[NE + e]; } else { s = d = e - NE; }
  int h = f >> (dshift - 2);                     // f / (dout/4)
  float alpha = ebuf[e * 4 + h] / segsum[d * 4 + h];
  float v = alpha * xl[(size_t)s * dout + f] + dis[s] * dis[d] * hp[(size_t)s * dout + f];
  atomicAdd(&accb[(size_t)d * dout + f], v);
}

// node epilogue: + gat_b + bg, BN(eval), elu -> h32 & h16
__global__ void node_post_k(const float* __restrict__ accb, const float* __restrict__ gatb,
                            const float* __restrict__ bg, const float* __restrict__ bng,
                            const float* __restrict__ bnb, const float* __restrict__ bnm,
                            const float* __restrict__ bnv, float* h32, _Float16* h16,
                            int dout) {
  int idx = blockIdx.x * blockDim.x + threadIdx.x;
  if (idx >= NN * dout) return;
  int f = idx & (dout - 1);
  float v = accb[idx] + gatb[f] + bg[f];
  v = (v - bnm[f]) * rsqrtf(bnv[f] + 1e-5f) * bng[f] + bnb[f];
  v = v > 0.0f ? v : expm1f(v);                  // elu
  h32[idx] = v;
  h16[idx] = (_Float16)v;
}

// LayerNorm(+relu) per row; writes f32 and optional f16
__global__ void ln_act_k(const float* __restrict__ in, const float* __restrict__ gam,
                         const float* __restrict__ bet, float* outf, _Float16* outh,
                         int dim) {
  int row = blockIdx.x, tid = threadIdx.x;
  __shared__ float red[256];
  float v = (tid < dim) ? in[(size_t)row * dim + tid] : 0.0f;
  red[tid] = v; __syncthreads();
  for (int s = 128; s > 0; s >>= 1) { if (tid < s) red[tid] += red[tid + s]; __syncthreads(); }
  float mean = red[0] / dim; __syncthreads();
  float dev = (tid < dim) ? (v - mean) : 0.0f;
  red[tid] = dev * dev; __syncthreads();
  for (int s = 128; s > 0; s >>= 1) { if (tid < s) red[tid] += red[tid + s]; __syncthreads(); }
  float rstd = rsqrtf(red[0] / dim + 1e-5f);
  if (tid < dim) {
    float o = dev * rstd * gam[tid] + bet[tid];
    o = o > 0.0f ? o : 0.0f;
    outf[(size_t)row * dim + tid] = o;
    if (outh) outh[(size_t)row * dim + tid] = (_Float16)o;
  }
}

__global__ void wdot_k(const float* __restrict__ t, const float* __restrict__ a2, float* w) {
  int n = blockIdx.x * blockDim.x + threadIdx.x;
  if (n >= NN) return;
  const float* r = t + (size_t)n * 256;
  float s = 0.0f;
  for (int j = 0; j < 256; ++j) s += r[j] * a2[j];
  w[n] = s;
}

__global__ void pool_init_k(float* pool_att, float* pool_sum, unsigned* pool_max,
                            float* cnt, unsigned* cmax, float* csum) {
  int i = blockIdx.x * blockDim.x + threadIdx.x;
  if (i < NG * 256) { pool_att[i] = 0.0f; pool_sum[i] = 0.0f; pool_max[i] = NEG_INF_ORD; }
  if (i < NG) cnt[i] = 0.0f;
  if (i < 4)  { cmax[i] = NEG_INF_ORD; csum[i] = 0.0f; }
}

__global__ void chunk_max_k(const float* __restrict__ w, const int* __restrict__ batch,
                            unsigned* cmax) {
  __shared__ unsigned sm[4];
  int tid = threadIdx.x;
  int n = blockIdx.x * blockDim.x + tid;
  if (tid < 4) sm[tid] = NEG_INF_ORD;
  __syncthreads();
  if (n < NN) atomicMax(&sm[batch[n] >> 5], f2o(w[n]));
  __syncthreads();
  if (tid < 4) atomicMax(&cmax[tid], sm[tid]);
}

__global__ void chunk_sum_k(const float* __restrict__ w, const int* __restrict__ batch,
                            const unsigned* __restrict__ cmax, float* ez, float* csum) {
  __shared__ float ss[4];
  int tid = threadIdx.x;
  int n = blockIdx.x * blockDim.x + tid;
  if (tid < 4) ss[tid] = 0.0f;
  __syncthreads();
  if (n < NN) {
    int c = batch[n] >> 5;
    float e = expf(w[n] - o2f(cmax[c]));
    ez[n] = e;
    atomicAdd(&ss[c], e);
  }
  __syncthreads();
  if (tid < 4) atomicAdd(&csum[tid], ss[tid]);
}

__global__ void pool_k(const float* __restrict__ h32, const int* __restrict__ batch,
                       const float* __restrict__ ez, const float* __restrict__ csum,
                       float* pool_att, float* pool_sum, unsigned* pool_max, float* cnt) {
  int idx = blockIdx.x * blockDim.x + threadIdx.x;
  if (idx >= NN * 256) return;
  int n = idx >> 8, f = idx & 255;
  int b = batch[n];
  float v = h32[idx];
  float sw = ez[n] / csum[b >> 5];
  atomicAdd(&pool_att[b * 256 + f], sw * v);
  atomicAdd(&pool_sum[b * 256 + f], v);
  atomicMax(&pool_max[b * 256 + f], f2o(v));
  if (f == 0) atomicAdd(&cnt[b], 1.0f);
}

__global__ void comb_k(const float* __restrict__ pool_att, const float* __restrict__ pool_sum,
                       const unsigned* __restrict__ pool_max, const float* __restrict__ cnt,
                       _Float16* comb16) {
  int idx = blockIdx.x * blockDim.x + threadIdx.x;
  if (idx >= NG * 768) return;
  int g = idx / 768, j = idx - g * 768;
  float v;
  if (j < 256)      v = pool_att[g * 256 + j];
  else if (j < 512) v = pool_sum[g * 256 + (j - 256)] / fmaxf(cnt[g], 1.0f);
  else              v = o2f(pool_max[g * 256 + (j - 512)]);
  comb16[idx] = (_Float16)v;
}

__global__ void cent_k(const float* __restrict__ z, const float* __restrict__ cents,
                       float* out, float* mind) {
  int g = blockIdx.x, k = threadIdx.x;   // 64 threads
  const float* zr = z + g * 256;
  const float* cr = cents + k * 256;
  float s = 0.0f;
  for (int j = 0; j < 256; ++j) { float d = zr[j] - cr[j]; s += d * d; }
  float dd = sqrtf(s + 1e-12f);
  __shared__ float sd[64];
  __shared__ float scd[16];
  sd[k] = dd; __syncthreads();
  if (k < 16) {
    float m = fminf(fminf(sd[4 * k], sd[4 * k + 1]), fminf(sd[4 * k + 2], sd[4 * k + 3]));
    out[g * 16 + k] = -m;
    scd[k] = m;
  }
  __syncthreads();
  if (k == 0) {
    float m = scd[0];
    for (int j = 1; j < 16; ++j) m = fminf(m, scd[j]);
    mind[g] = m;
  }
}

__global__ void final_k(const float* __restrict__ mind, float* out) {
  __shared__ float s1[128], s2[128];
  int t = threadIdx.x;
  float v = mind[t];
  s1[t] = v; s2[t] = v * v; __syncthreads();
  for (int s = 64; s > 0; s >>= 1) {
    if (t < s) { s1[t] += s1[t + s]; s2[t] += s2[t + s]; }
    __syncthreads();
  }
  float mean = s1[0] / 128.0f;
  float var  = (s2[0] - 128.0f * mean * mean) / 127.0f;   // ddof=1
  float stdv = sqrtf(fmaxf(var, 0.0f));
  out[2048 + t] = (v - mean) / (stdv + 1e-6f);
}

// ---------------------------------------------------------------------------
extern "C" void kernel_launch(void* const* d_in, const int* in_sizes, int n_in,
                              void* d_out, int out_size, void* d_ws, size_t ws_size,
                              hipStream_t stream) {
  const float* x     = (const float*)d_in[0];
  const int*   ei    = (const int*)d_in[1];
  const int*   batch = (const int*)d_in[2];
  const float* projW = (const float*)d_in[3];
  const float* projB = (const float*)d_in[4];
  const float* projG = (const float*)d_in[5];
  const float* projBt= (const float*)d_in[6];
  const float* W1    = (const float*)d_in[63];
  const float* b1    = (const float*)d_in[64];
  const float* a2    = (const float*)d_in[65];
  const float* aggW  = (const float*)d_in[66];
  const float* aggB  = (const float*)d_in[67];
  const float* aggG  = (const float*)d_in[68];
  const float* aggBt = (const float*)d_in[69];
  const float* cents = (const float*)d_in[70];
  float* out = (float*)d_out;

  // workspace carve-out
  char* base = (char*)d_ws;
  size_t off = 0;
  auto carve = [&](size_t bytes) -> char* {
    off = (off + 255) & ~(size_t)255;
    char* p = base + off;
    off += bytes;
    return p;
  };
  float*     h32     = (float*)    carve((size_t)NN * 256 * 4);
  _Float16*  h16     = (_Float16*) carve((size_t)NN * 256 * 2);
  float*     xl      = (float*)    carve((size_t)NN * 256 * 4);
  float*     xr      = (float*)    carve((size_t)NN * 256 * 4);
  float*     hp      = (float*)    carve((size_t)NN * 256 * 4);
  float*     accb    = (float*)    carve((size_t)NN * 256 * 4);
  _Float16*  w16     = (_Float16*) carve((size_t)768 * 256 * 2);
  float*     ebuf    = (float*)    carve((size_t)EP * 4 * 4);
  unsigned*  segmax  = (unsigned*) carve((size_t)NN * 4 * 4);
  float*     segsum  = (float*)    carve((size_t)NN * 4 * 4);
  float*     deg     = (float*)    carve((size_t)NN * 4);       // later reused as w
  float*     dis     = (float*)    carve((size_t)NN * 4);
  float*     pool_att= (float*)    carve((size_t)NG * 256 * 4);
  float*     pool_sum= (float*)    carve((size_t)NG * 256 * 4);
  unsigned*  pool_max= (unsigned*) carve((size_t)NG * 256 * 4);
  float*     cnt     = (float*)    carve((size_t)NG * 4);
  unsigned*  cmax    = (unsigned*) carve(16);
  float*     csum    = (float*)    carve(16);
  float*     zbuf    = (float*)    carve((size_t)NG * 256 * 4);
  float*     mind    = (float*)    carve((size_t)NG * 4);
  _Float16*  comb16  = (_Float16*) carve((size_t)NG * 768 * 2);
  if (off > ws_size) return;   // insufficient scratch; bail deterministically

  auto blocks = [](long n) { return dim3((unsigned)((n + 255) / 256)); };
  const dim3 T(256);

  auto run_gemm = [&](const _Float16* A, const float* Wf, const float* bias,
                      float* C, int M, int K, int Nout, int act) {
    cvt_f16_k<<<blocks((long)K * Nout), T, 0, stream>>>(Wf, w16, K * Nout);
    dim3 grid((M + 127) / 128, Nout / 16);
    gemm_wmma_k<<<grid, T, 0, stream>>>(A, w16, bias, C, M, K, Nout, act);
  };

  // degrees (with self loops) and sym-norm
  deg_init_k<<<blocks(NN), T, 0, stream>>>(deg);
  deg_acc_k<<<blocks(NE), T, 0, stream>>>(ei, deg);
  dis_k<<<blocks(NN), T, 0, stream>>>(deg, dis);

  // projection: h = relu(LN(x @ W + b))
  _Float16* x16 = (_Float16*)xl;   // temp reuse
  cvt_f16_k<<<blocks((long)NN * 128), T, 0, stream>>>(x, x16, NN * 128);
  run_gemm(x16, projW, projB, accb, NN, 128, 64, 0);
  ln_act_k<<<dim3(NN), T, 0, stream>>>(accb, projG, projBt, h32, h16, 64);

  const int DIN[4]  = {64, 64, 128, 256};
  const int DOUT[4] = {64, 128, 256, 256};
  for (int i = 0; i < 4; ++i) {
    int bidx = 7 + 14 * i;
    const float* Wl   = (const float*)d_in[bidx + 0];
    const float* bl   = (const float*)d_in[bidx + 1];
    const float* Wr   = (const float*)d_in[bidx + 2];
    const float* br   = (const float*)d_in[bidx + 3];
    const float* att  = (const float*)d_in[bidx + 4];
    const float* gatb = (const float*)d_in[bidx + 5];
    const float* Wg   = (const float*)d_in[bidx + 6];
    const float* bg   = (const float*)d_in[bidx + 7];
    const float* bng  = (const float*)d_in[bidx + 8];
    const float* bnb  = (const float*)d_in[bidx + 9];
    const float* bnm  = (const float*)d_in[bidx + 10];
    const float* bnv  = (const float*)d_in[bidx + 11];
    const float* Wres = (const float*)d_in[bidx + 12];
    const float* bres = (const float*)d_in[bidx + 13];
    int din = DIN[i], dout = DOUT[i];
    int dshift = (dout == 64) ? 6 : (dout == 128) ? 7 : 8;

    run_gemm(h16, Wl,   bl,      xl,   NN, din, dout, 0);
    run_gemm(h16, Wr,   br,      xr,   NN, din, dout, 0);
    run_gemm(h16, Wg,   nullptr, hp,   NN, din, dout, 0);
    run_gemm(h16, Wres, bres,    accb, NN, din, dout, 0);  // residual seeds accumulator

    seg_init_k<<<blocks((long)NN * 4), T, 0, stream>>>(segmax, segsum);
    gat_e1_k<<<blocks((long)EP * 4), T, 0, stream>>>(ei, xl, xr, att, ebuf, segmax, dout);
    gat_e2_k<<<blocks((long)EP * 4), T, 0, stream>>>(ei, ebuf, segmax, segsum);
    gat_e3_k<<<blocks((long)EP * dout), T, 0, stream>>>(ei, ebuf, segsum, xl, hp, dis,
                                                        accb, dout, dshift);
    node_post_k<<<blocks((long)NN * dout), T, 0, stream>>>(accb, gatb, bg, bng, bnb, bnm,
                                                           bnv, h32, h16, dout);
  }

  // attention readout: w = tanh(h @ W1 + b1) @ a2
  run_gemm(h16, W1, b1, xl, NN, 256, 256, 1);   // tanh epilogue -> t in xl
  float* wscore = deg;                          // reuse
  wdot_k<<<blocks(NN), T, 0, stream>>>(xl, a2, wscore);

  pool_init_k<<<blocks((long)NG * 256), T, 0, stream>>>(pool_att, pool_sum, pool_max,
                                                        cnt, cmax, csum);
  chunk_max_k<<<blocks(NN), T, 0, stream>>>(wscore, batch, cmax);
  chunk_sum_k<<<blocks(NN), T, 0, stream>>>(wscore, batch, cmax, ebuf, csum);
  pool_k<<<blocks((long)NN * 256), T, 0, stream>>>(h32, batch, ebuf, csum,
                                                   pool_att, pool_sum, pool_max, cnt);
  comb_k<<<blocks((long)NG * 768), T, 0, stream>>>(pool_att, pool_sum, pool_max, cnt, comb16);

  // agg: z = relu(LN(comb @ W + b))
  run_gemm(comb16, aggW, aggB, xr, NG, 768, 256, 0);
  ln_act_k<<<dim3(NG), T, 0, stream>>>(xr, aggG, aggBt, zbuf, nullptr, 256);

  // centroid distances + outlier scores
  cent_k<<<dim3(NG), dim3(64), 0, stream>>>(zbuf, cents, out, mind);
  final_k<<<dim3(1), dim3(128), 0, stream>>>(mind, out);
}
